// SelfAttention_76390288327204
// MI455X (gfx1250) — compile-verified, matrix operand order checked by reading, not run on hardware
//
#include <hip/hip_runtime.h>

typedef __attribute__((ext_vector_type(16))) __bf16 v16bf;
typedef __attribute__((ext_vector_type(8)))  float  v8f;

struct __attribute__((aligned(16))) U128 { unsigned int u[4]; };
union FragAB { v16bf v; U128 q[2]; };

#define EMBED   1024
#define HEADS   16
#define HD      64
#define SEQL    2048
#define BATCH   2
#define QKV_LD  3072

// ---- TDM availability (6-arg clang-23 form), with manual-copy fallback ----
#if defined(__has_builtin)
#  if __has_builtin(__builtin_amdgcn_tensor_load_to_lds)
#    define USE_TDM 1
#  endif
#endif
#ifndef USE_TDM
#  define USE_TDM 0
#endif

typedef unsigned int tdm_u4 __attribute__((ext_vector_type(4)));
typedef int          tdm_i8 __attribute__((ext_vector_type(8)));
typedef int          tdm_i4 __attribute__((ext_vector_type(4)));

__device__ __forceinline__ unsigned lds_offset_of(const void* p) {
  // low 32 bits of a generic LDS address == wave-relative LDS byte offset
  return (unsigned)(unsigned long long)p;
}

// xor-butterfly lane exchange via ds_swizzle (group-of-32, and_mask=0x1f):
// no address VGPR setup, stays within the 16-lane C-fragment group for XORM<16
template <int XORM>
__device__ __forceinline__ float swz_xor(float x) {
  int r = __builtin_amdgcn_ds_swizzle(__float_as_int(x), 0x1f | (XORM << 10));
  return __int_as_float(r);
}
__device__ __forceinline__ float max16(float x) {
  x = fmaxf(x, swz_xor<1>(x));
  x = fmaxf(x, swz_xor<2>(x));
  x = fmaxf(x, swz_xor<4>(x));
  x = fmaxf(x, swz_xor<8>(x));
  return x;
}

// ---------------------------------------------------------------------------
// f32 -> bf16 conversion (grid-stride)
// ---------------------------------------------------------------------------
__global__ void cvt_f32_bf16(const float* __restrict__ in,
                             __bf16* __restrict__ out, int n) {
  for (int i = blockIdx.x * blockDim.x + threadIdx.x; i < n;
       i += gridDim.x * blockDim.x)
    out[i] = (__bf16)in[i];
}

// ---------------------------------------------------------------------------
// bf16 WMMA GEMM: C[M,N] = A[M,K] * B[K,N]  (A,B row-major bf16)
// Block 128x128x32, 8 wave32 waves (2x4), wave tile 64x32 (4x2 WMMA tiles).
// A tile staged via TDM tensor_load_to_lds (wave 0 issues descriptor, rows
// padded 64B+16B to match PAD_K); B staged transposed by all threads.
// ---------------------------------------------------------------------------
#define BM 128
#define BN 128
#define BK 32
#define PAD_K 40   // padded row length (bf16 elems): 80B rows, 16B aligned

template <bool F32OUT>
__global__ __launch_bounds__(256) void gemm_bf16(
    const __bf16* __restrict__ A, const __bf16* __restrict__ B,
    void* __restrict__ Cout, int M, int N, int K)
{
  __shared__ __bf16 sA[BM * PAD_K];
  __shared__ __bf16 sB[BN * PAD_K];   // transposed: [n][k]

  const int tid   = threadIdx.x;
  const int lane  = tid & 31;
  const int wid   = tid >> 5;
  const int waveM = wid >> 2;         // 0..1
  const int waveN = wid & 3;          // 0..3
  const int lhalf = lane >> 4;        // 0/1
  const int l16   = lane & 15;
  const int kh    = lhalf ? 8 : 0;    // K-half base for A/B fragments

  const int rowBase = blockIdx.y * BM;
  const int colBase = blockIdx.x * BN;

  v8f acc[4][2];
  #pragma unroll
  for (int m = 0; m < 4; ++m)
    #pragma unroll
    for (int n = 0; n < 2; ++n)
      #pragma unroll
      for (int j = 0; j < 8; ++j) acc[m][n][j] = 0.f;

#if USE_TDM
  const unsigned sA_lds = lds_offset_of(&sA[0]);
#endif

  for (int k0 = 0; k0 < K; k0 += BK) {
    // ---- stage A tile -----------------------------------------------------
#if USE_TDM
    if (wid == 0) {
      unsigned long long ga =
          (unsigned long long)(const void*)(A + (size_t)rowBase * K + k0);
      tdm_u4 g0;
      g0[0] = 1u;                                   // count=1, user mode
      g0[1] = sA_lds;                               // lds_addr
      g0[2] = (unsigned)(ga & 0xffffffffu);         // global_addr[31:0]
      g0[3] = (unsigned)((ga >> 32) & 0x01ffffffu)  // global_addr[56:32]
              | 0x80000000u;                        // type=2 ("image")
      tdm_i8 g1;
      g1[0] = (1 << 16)    // data_size = 2 bytes
            | (1 << 20)    // pad_enable
            | (3 << 22)    // pad_interval: 16 DWORDs (one 64B tile row)
            | (3 << 25);   // pad_amount:   4 DWORDs (16B -> PAD_K=40)
      g1[1] = (int)((K & 0xffff) << 16);                        // tensor_dim0 lo
      g1[2] = (int)(((unsigned)K >> 16) | ((M & 0xffff) << 16));// td0 hi, td1 lo
      g1[3] = (int)(((unsigned)M >> 16) | (BK << 16));          // td1 hi, tile_dim0=32
      g1[4] = BM;                                               // tile_dim1=128
      g1[5] = K;                                                // dim0 stride lo32
      g1[6] = 0;
      g1[7] = 0;
      tdm_i4 g2 = {0, 0, 0, 0};
      tdm_i4 g3 = {0, 0, 0, 0};
      tdm_i8 g4 = {0, 0, 0, 0, 0, 0, 0, 0};
      __builtin_amdgcn_tensor_load_to_lds(g0, g1, g2, g3, g4, 0);
    }
#else
    {
      int r = tid >> 2;                // 0..63
      int c = (tid & 3) * 8;           // 0,8,16,24
      #pragma unroll
      for (int it = 0; it < 2; ++it, r += 64) {
        U128 d = *(const U128*)(A + (size_t)(rowBase + r) * K + k0 + c);
        *(U128*)(&sA[r * PAD_K + c]) = d;
      }
    }
#endif
    // ---- stage B tile transposed -----------------------------------------
    {
      int k = tid >> 4;                // 0..15
      int n = (tid & 15) * 8;
      #pragma unroll
      for (int it = 0; it < 2; ++it, k += 16) {
        U128 d = *(const U128*)(B + (size_t)(k0 + k) * N + colBase + n);
        const __bf16* e = (const __bf16*)&d;
        #pragma unroll
        for (int i = 0; i < 8; ++i) sB[(n + i) * PAD_K + k] = e[i];
        if (k0 + BK < K)
          __builtin_prefetch(B + (size_t)(k0 + BK + k) * N + colBase + n, 0, 0);
      }
    }
#if USE_TDM
    if (wid == 0) __builtin_amdgcn_s_wait_tensorcnt(0);
#endif
    __syncthreads();

    FragAB af[4], bfr[2];
    #pragma unroll
    for (int m = 0; m < 4; ++m) {
      const __bf16* p = &sA[(waveM * 64 + m * 16 + l16) * PAD_K];
      af[m].q[0] = *(const U128*)(p + kh);        // K = kh..kh+7
      af[m].q[1] = *(const U128*)(p + kh + 16);   // K = kh+16..kh+23
    }
    #pragma unroll
    for (int n = 0; n < 2; ++n) {
      const __bf16* p = &sB[(waveN * 32 + n * 16 + l16) * PAD_K];
      bfr[n].q[0] = *(const U128*)(p + kh);
      bfr[n].q[1] = *(const U128*)(p + kh + 16);
    }
    #pragma unroll
    for (int m = 0; m < 4; ++m)
      #pragma unroll
      for (int n = 0; n < 2; ++n)
        acc[m][n] = __builtin_amdgcn_wmma_f32_16x16x32_bf16(
            false, af[m].v, false, bfr[n].v, (short)0, acc[m][n], false, false);
    __syncthreads();
  }

  // epilogue: C-layout  M = 8*lhalf + vgpr,  N = l16  (branch-free)
  const size_t cbase = (size_t)(rowBase + waveM * 64 + lhalf * 8) * N
                     + colBase + waveN * 32 + l16;
  #pragma unroll
  for (int m = 0; m < 4; ++m)
    #pragma unroll
    for (int n = 0; n < 2; ++n)
      #pragma unroll
      for (int j = 0; j < 8; ++j) {
        size_t idx = cbase + (size_t)(m * 16 + j) * N + n * 16;
        float v = acc[m][n][j];
        if constexpr (F32OUT) ((float*)Cout)[idx] = v;
        else                  ((__bf16*)Cout)[idx] = (__bf16)v;
      }
}

// ---------------------------------------------------------------------------
// T5 relative position bucket (num_buckets=32, max_distance=128)
// ---------------------------------------------------------------------------
__device__ __forceinline__ int t5_bucket(int rel) {  // rel = k - q
  int rb = (rel > 0) ? 16 : 0;
  int a  = rel < 0 ? -rel : rel;
  if (a < 8) return rb + a;
  int lg = (int)(__logf((float)a * 0.125f) * 2.8853900817779268f);
  lg = lg > 7 ? 7 : lg;
  return rb + 8 + lg;
}

// ---------------------------------------------------------------------------
// Flash attention: block per (qblock=128, head, batch); 8 waves x 16 q rows.
// KV walked in chunks of 32.  K tile staged per chunk via
// global_load_async_to_lds_b128 (ASYNCcnt); V staged transposed; bias via
// per-head LDS table; row max via ds_swizzle butterfly; row sum via a
// P x ones WMMA (sum lands in the same C-fragment layout as the O acc).
// ---------------------------------------------------------------------------
__global__ __launch_bounds__(256) void attn_fwd(
    const __bf16* __restrict__ qkv, const float* __restrict__ rel_emb,
    __bf16* __restrict__ outb)
{
  __shared__ __bf16 sK[32 * 72];        // K tile [kv][d], padded rows (144B)
  __shared__ __bf16 sVT[HD * 40];       // V^T tile [d][kv]
  __shared__ __bf16 sP[8][16 * 40];     // per-wave P scratch (16x32)
  __shared__ float  sBias[4096];        // bias by (rel+2048) for this head

  const int tid   = threadIdx.x;
  const int lane  = tid & 31;
  const int wid   = tid >> 5;
  const int lhalf = lane >> 4;
  const int l16   = lane & 15;
  const int kh    = lhalf ? 8 : 0;

  const int h     = blockIdx.y;
  const int b     = blockIdx.z;
  const int qbase = blockIdx.x * 128 + wid * 16;

  // per-head bias table: bias(rel) = rel_emb[bucket(rel)*H + h]
  for (int i = tid; i < 4096; i += 256)
    sBias[i] = rel_emb[t5_bucket(i - 2048) * HEADS + h];

  const size_t bOff = (size_t)b * SEQL * QKV_LD;
  const __bf16* qb = qkv + bOff + h * HD;
  const __bf16* kb = qkv + bOff + EMBED + h * HD;
  const __bf16* vb = qkv + bOff + 2 * EMBED + h * HD;

  // Q fragments (A-layout), 2 K-chunks over D=64
  FragAB aq[2];
  {
    const __bf16* qp = qb + (size_t)(qbase + l16) * QKV_LD;
    #pragma unroll
    for (int c = 0; c < 2; ++c) {
      aq[c].q[0] = *(const U128*)(qp + c * 32 + kh);
      aq[c].q[1] = *(const U128*)(qp + c * 32 + kh + 16);
    }
  }

  // all-ones B fragment: row-sum WMMA operand
  FragAB ones;
  #pragma unroll
  for (int i = 0; i < 16; ++i) ones.v[i] = (__bf16)1.0f;

  v8f acc[4];
  float mrow[8], lrow[8], corr[8];
  #pragma unroll
  for (int t = 0; t < 4; ++t)
    #pragma unroll
    for (int j = 0; j < 8; ++j) acc[t][j] = 0.f;
  #pragma unroll
  for (int j = 0; j < 8; ++j) { mrow[j] = -1e30f; lrow[j] = 0.f; }

  const int skv = tid >> 3;            // 0..31  (staging row)
  const int sd  = (tid & 7) * 8;       // 0..56  (staging dim)

  for (int kv0 = 0; kv0 < SEQL; kv0 += 32) {
    // ---- stage K tile via async copy to LDS (one b128 per thread) --------
    {
      const __bf16* g = kb + (size_t)(kv0 + skv) * QKV_LD + sd;
      unsigned lo = lds_offset_of(&sK[skv * 72 + sd]);
      asm volatile("global_load_async_to_lds_b128 %0, %1, off"
                   :: "v"(lo), "v"(g) : "memory");
    }
    // ---- stage V tile transposed -----------------------------------------
    {
      U128 dd = *(const U128*)(vb + (size_t)(kv0 + skv) * QKV_LD + sd);
      const __bf16* e = (const __bf16*)&dd;
      #pragma unroll
      for (int i = 0; i < 8; ++i) sVT[(sd + i) * 40 + skv] = e[i];
    }
    if (kv0 + 32 < SEQL) {
      __builtin_prefetch(kb + (size_t)(kv0 + 32 + skv) * QKV_LD + sd, 0, 0);
      __builtin_prefetch(vb + (size_t)(kv0 + 32 + skv) * QKV_LD + sd, 0, 0);
    }
    asm volatile("s_wait_asynccnt 0x0" ::: "memory");
    __syncthreads();   // sK, sVT (and first-iter sBias) ready

    // ---- scores: S = Q * K^T, two 16-col tiles ---------------------------
    v8f s0, s1;
    #pragma unroll
    for (int j = 0; j < 8; ++j) { s0[j] = 0.f; s1[j] = 0.f; }
    #pragma unroll
    for (int c = 0; c < 2; ++c) {
      FragAB bk0, bk1;
      const __bf16* kp  = &sK[l16 * 72 + c * 32];
      const __bf16* kp1 = &sK[(16 + l16) * 72 + c * 32];
      bk0.q[0] = *(const U128*)(kp + kh);
      bk0.q[1] = *(const U128*)(kp + kh + 16);
      bk1.q[0] = *(const U128*)(kp1 + kh);
      bk1.q[1] = *(const U128*)(kp1 + kh + 16);
      s0 = __builtin_amdgcn_wmma_f32_16x16x32_bf16(
          false, aq[c].v, false, bk0.v, (short)0, s0, false, false);
      s1 = __builtin_amdgcn_wmma_f32_16x16x32_bf16(
          false, aq[c].v, false, bk1.v, (short)0, s1, false, false);
    }

    // ---- bias + online softmax (C-layout: row = 8*lhalf+j, col = l16) ----
    #pragma unroll
    for (int j = 0; j < 8; ++j) {
      int rel = kv0 + l16 - (qbase + lhalf * 8 + j) + 2048;
      float v0 = s0[j] + sBias[rel];
      float v1 = s1[j] + sBias[rel + 16];
      float mnew = fmaxf(mrow[j], max16(fmaxf(v0, v1)));
      float p0 = __expf(v0 - mnew);
      float p1 = __expf(v1 - mnew);
      corr[j] = __expf(mrow[j] - mnew);
      mrow[j] = mnew;
      #pragma unroll
      for (int t = 0; t < 4; ++t) acc[t][j] *= corr[j];
      sP[wid][(lhalf * 8 + j) * 40 + l16]      = (__bf16)p0;
      sP[wid][(lhalf * 8 + j) * 40 + 16 + l16] = (__bf16)p1;
    }

    // ---- O += P * V ; row sums via P x ones (same-wave LDS reload) -------
    FragAB ap;
    {
      const __bf16* pp = &sP[wid][l16 * 40];
      ap.q[0] = *(const U128*)(pp + kh);
      ap.q[1] = *(const U128*)(pp + kh + 16);
    }
    v8f zc;
    #pragma unroll
    for (int j = 0; j < 8; ++j) zc[j] = 0.f;
    v8f psum = __builtin_amdgcn_wmma_f32_16x16x32_bf16(
        false, ap.v, false, ones.v, (short)0, zc, false, false);
    #pragma unroll
    for (int j = 0; j < 8; ++j) lrow[j] = lrow[j] * corr[j] + psum[j];

    #pragma unroll
    for (int t = 0; t < 4; ++t) {
      FragAB bv;
      const __bf16* vp = &sVT[(t * 16 + l16) * 40];
      bv.q[0] = *(const U128*)(vp + kh);
      bv.q[1] = *(const U128*)(vp + kh + 16);
      acc[t] = __builtin_amdgcn_wmma_f32_16x16x32_bf16(
          false, ap.v, false, bv.v, (short)0, acc[t], false, false);
    }
    __syncthreads();   // everyone done with sK/sVT/sP before next stage
  }

  // ---- normalize + store attn-out in (B,S,E) bf16 ------------------------
  const size_t oOff = (size_t)b * SEQL * EMBED;
  #pragma unroll
  for (int t = 0; t < 4; ++t)
    #pragma unroll
    for (int j = 0; j < 8; ++j) {
      float v = acc[t][j] / lrow[j];
      int row = qbase + lhalf * 8 + j;
      int col = h * HD + t * 16 + l16;
      outb[oOff + (size_t)row * EMBED + col] = (__bf16)v;
    }
}

// ---------------------------------------------------------------------------
extern "C" void kernel_launch(void* const* d_in, const int* in_sizes, int n_in,
                              void* d_out, int out_size, void* d_ws, size_t ws_size,
                              hipStream_t stream) {
  const float* hidden  = (const float*)d_in[0];
  const float* Wqkv    = (const float*)d_in[1];
  const float* Wout    = (const float*)d_in[2];
  const float* rel_emb = (const float*)d_in[3];

  char* p = (char*)d_ws;
  __bf16* hB    = (__bf16*)p; p += (size_t)BATCH * SEQL * EMBED * 2;      //  8 MB
  __bf16* WqkvB = (__bf16*)p; p += (size_t)EMBED * 3 * EMBED * 2;         //  6 MB
  __bf16* WoutB = (__bf16*)p; p += (size_t)EMBED * EMBED * 2;             //  2 MB
  __bf16* qkvB  = (__bf16*)p; p += (size_t)BATCH * SEQL * 3 * EMBED * 2;  // 24 MB
  __bf16* attnB = (__bf16*)p;                                             //  8 MB

  cvt_f32_bf16<<<2048, 256, 0, stream>>>(hidden, hB, BATCH * SEQL * EMBED);
  cvt_f32_bf16<<<2048, 256, 0, stream>>>(Wqkv, WqkvB, EMBED * 3 * EMBED);
  cvt_f32_bf16<<<1024, 256, 0, stream>>>(Wout, WoutB, EMBED * EMBED);

  // QKV projection: [4096,1024] x [1024,3072] -> bf16 per-head qkv buffer
  gemm_bf16<false><<<dim3(3 * EMBED / BN, BATCH * SEQL / BM), 256, 0, stream>>>(
      hB, WqkvB, qkvB, BATCH * SEQL, 3 * EMBED, EMBED);

  // fused bias + softmax + PV
  attn_fwd<<<dim3(SEQL / 128, HEADS, BATCH), 256, 0, stream>>>(
      qkvB, rel_emb, attnB);

  // output projection: [4096,1024] x [1024,1024] -> fp32 d_out
  gemm_bf16<true><<<dim3(EMBED / BN, BATCH * SEQL / BM), 256, 0, stream>>>(
      attnB, WoutB, d_out, BATCH * SEQL, EMBED, EMBED);
}
